// OutlierAwareLinear_70944269795991
// MI455X (gfx1250) — matrix operand order repaired and checked by reading, not compiled, not run on hardware
//
#include <hip/hip_runtime.h>

typedef __attribute__((ext_vector_type(16))) __bf16 v16bf;
typedef __attribute__((ext_vector_type(8)))  float  v8f;

#define TILE_M 128
#define TILE_N 128
#define TILE_K 32

// Split one float into bf16 hi (truncate) and bf16 lo (residual), packing two
// K-adjacent elements per uint (low half = even K).
__device__ __forceinline__ void cvt2_hilo(float a, float b, unsigned& hi, unsigned& lo) {
    unsigned ua = __float_as_uint(a);
    unsigned ha = ua >> 16;
    float    ra = a - __uint_as_float(ha << 16);
    unsigned la = __float_as_uint(ra) >> 16;
    unsigned ub = __float_as_uint(b);
    unsigned hb = ub >> 16;
    float    rb = b - __uint_as_float(hb << 16);
    unsigned lb = __float_as_uint(rb) >> 16;
    hi = ha | (hb << 16);
    lo = la | (lb << 16);
}

union FragBF16 {
    v16bf v;
    uint4 q[2];
};

__global__ __launch_bounds__(256)
void oal_gemm_bf16x3_wmma(const float* __restrict__ x,
                          const float* __restrict__ W,
                          const float* __restrict__ bias,
                          float* __restrict__ y,
                          int M, int N, int K) {
    // Double-buffered LDS tiles, packed pairs of bf16 (uint = {K+1, K});
    // K-stride = 16 uints. 4 arrays x 2 buffers x 8 KB = 64 KB.
    __shared__ unsigned sAhi[2][TILE_M * TILE_K / 2];
    __shared__ unsigned sAlo[2][TILE_M * TILE_K / 2];
    __shared__ unsigned sBhi[2][TILE_N * TILE_K / 2];
    __shared__ unsigned sBlo[2][TILE_N * TILE_K / 2];

    const int tid   = threadIdx.x;
    const int lane  = tid & 31;
    const int wave  = tid >> 5;       // 0..7
    const int waveM = wave >> 1;      // 0..3 -> 32-row slab
    const int waveN = wave & 1;       // 0..1 -> 64-col slab
    const int lr      = lane & 15;    // row/col within 16
    const int halfSel = lane >> 4;    // K-half selector per ISA layout

    const int rowBase = blockIdx.y * TILE_M;
    const int colBase = blockIdx.x * TILE_N;

    // Cooperative staging: both tiles are 128 rows x 32 K fp32
    // (4096 floats = 16 floats/thread each).
    const int sRow = tid >> 1;            // 0..127
    const int sK   = (tid & 1) * 16;      // 0 or 16
    const int sIdx = sRow * 16 + (sK >> 1);

    const float* aG = x + (size_t)(rowBase + sRow) * K + sK;
    const float* bG = W + (size_t)(colBase + sRow) * K + sK;

    v8f acc[2][4] = {};

    // ---- prologue: stage tile 0 into buffer 0 ----
    {
        const float4* ap = (const float4*)(aG);
        const float4* bp = (const float4*)(bG);
#pragma unroll
        for (int i = 0; i < 4; ++i) {
            float4 fa = ap[i];
            unsigned h0, l0, h1, l1;
            cvt2_hilo(fa.x, fa.y, h0, l0);
            cvt2_hilo(fa.z, fa.w, h1, l1);
            sAhi[0][sIdx + i * 2] = h0; sAhi[0][sIdx + i * 2 + 1] = h1;
            sAlo[0][sIdx + i * 2] = l0; sAlo[0][sIdx + i * 2 + 1] = l1;
            float4 fb = bp[i];
            cvt2_hilo(fb.x, fb.y, h0, l0);
            cvt2_hilo(fb.z, fb.w, h1, l1);
            sBhi[0][sIdx + i * 2] = h0; sBhi[0][sIdx + i * 2 + 1] = h1;
            sBlo[0][sIdx + i * 2] = l0; sBlo[0][sIdx + i * 2 + 1] = l1;
        }
    }
    __syncthreads();

    int buf = 0;
    for (int k0 = 0; k0 < K; k0 += TILE_K, buf ^= 1) {
        const bool hasNext = (k0 + TILE_K) < K;

        // ---- 1) issue next K-slab's global loads early (latency hidden by WMMAs) ----
        float4 ra[4], rb[4];
        if (hasNext) {
            const float4* ap = (const float4*)(aG + k0 + TILE_K);
            const float4* bp = (const float4*)(bG + k0 + TILE_K);
#pragma unroll
            for (int i = 0; i < 4; ++i) { ra[i] = ap[i]; rb[i] = bp[i]; }
        }

        // ---- 2) compute current tile from LDS buffer `buf` ----
        // A 16x32 layout: lanes 0-15 hold K 0-7 & 16-23; lanes 16-31 hold K 8-15 & 24-31
        FragBF16 ahi[2], alo[2];
#pragma unroll
        for (int t = 0; t < 2; ++t) {
            int row  = waveM * 32 + t * 16 + lr;
            int base = row * 16 + halfSel * 4;                 // uint index
            ahi[t].q[0] = *(const uint4*)&sAhi[buf][base];
            ahi[t].q[1] = *(const uint4*)&sAhi[buf][base + 8];
            alo[t].q[0] = *(const uint4*)&sAlo[buf][base];
            alo[t].q[1] = *(const uint4*)&sAlo[buf][base + 8];
        }
        // B 32x16 layout: lanes 0-15 hold K 0-7 & 16-23; lanes 16-31 hold K 8-15 & 24-31
#pragma unroll
        for (int j = 0; j < 4; ++j) {
            FragBF16 bhi, blo;
            int n    = waveN * 64 + j * 16 + lr;
            int base = n * 16 + halfSel * 4;
            bhi.q[0] = *(const uint4*)&sBhi[buf][base];
            bhi.q[1] = *(const uint4*)&sBhi[buf][base + 8];
            blo.q[0] = *(const uint4*)&sBlo[buf][base];
            blo.q[1] = *(const uint4*)&sBlo[buf][base + 8];
#pragma unroll
            for (int i = 0; i < 2; ++i) {
                // bf16x3: hi*hi + lo*hi + hi*lo
                acc[i][j] = __builtin_amdgcn_wmma_f32_16x16x32_bf16(
                    false, ahi[i].v, false, bhi.v, (short)0, acc[i][j], false, false);
                acc[i][j] = __builtin_amdgcn_wmma_f32_16x16x32_bf16(
                    false, alo[i].v, false, bhi.v, (short)0, acc[i][j], false, false);
                acc[i][j] = __builtin_amdgcn_wmma_f32_16x16x32_bf16(
                    false, ahi[i].v, false, blo.v, (short)0, acc[i][j], false, false);
            }
        }

        // ---- 3) convert staged registers, store into buffer `buf^1` ----
        if (hasNext) {
            const int nb = buf ^ 1;
#pragma unroll
            for (int i = 0; i < 4; ++i) {
                unsigned h0, l0, h1, l1;
                cvt2_hilo(ra[i].x, ra[i].y, h0, l0);
                cvt2_hilo(ra[i].z, ra[i].w, h1, l1);
                sAhi[nb][sIdx + i * 2] = h0; sAhi[nb][sIdx + i * 2 + 1] = h1;
                sAlo[nb][sIdx + i * 2] = l0; sAlo[nb][sIdx + i * 2 + 1] = l1;
                cvt2_hilo(rb[i].x, rb[i].y, h0, l0);
                cvt2_hilo(rb[i].z, rb[i].w, h1, l1);
                sBhi[nb][sIdx + i * 2] = h0; sBhi[nb][sIdx + i * 2 + 1] = h1;
                sBlo[nb][sIdx + i * 2] = l0; sBlo[nb][sIdx + i * 2 + 1] = l1;
            }
        }
        __syncthreads();   // next buffer ready; current buffer free for overwrite
    }

    // ---- epilogue: add bias, store fp32 ----
    // C/D layout: VGPR r -> (M = halfSel*8 + r, N = lane&15)
#pragma unroll
    for (int i = 0; i < 2; ++i) {
#pragma unroll
        for (int j = 0; j < 4; ++j) {
            int gCol = colBase + waveN * 64 + j * 16 + lr;
            float bv = bias[gCol];
#pragma unroll
            for (int r = 0; r < 8; ++r) {
                int gRow = rowBase + waveM * 32 + i * 16 + halfSel * 8 + r;
                y[(size_t)gRow * N + gCol] = acc[i][j][r] + bv;
            }
        }
    }
}

// Per-token outlier-aware fake quantization, in place. One block per token.
__global__ __launch_bounds__(256)
void oal_quant_rows(float* __restrict__ y, int D) {
    const int t = threadIdx.x;
    float* row = y + (size_t)blockIdx.x * D;

    // 2048 / 256 = 8 elements per thread, held in registers.
    const float4* p = (const float4*)(row + t * 8);
    float4 f0 = p[0];
    float4 f1 = p[1];
    float v[8] = {f0.x, f0.y, f0.z, f0.w, f1.x, f1.y, f1.z, f1.w};

    float s = 0.f, ss = 0.f;
#pragma unroll
    for (int i = 0; i < 8; ++i) { s += v[i]; ss += v[i] * v[i]; }

    __shared__ float r1[256];
    __shared__ float r2[256];
    r1[t] = s; r2[t] = ss;
    __syncthreads();
    for (int off = 128; off > 0; off >>= 1) {
        if (t < off) { r1[t] += r1[t + off]; r2[t] += r2[t + off]; }
        __syncthreads();
    }
    const float invD = 1.0f / (float)D;
    float mean = r1[0] * invD;
    float var  = r2[0] * invD - mean * mean;
    float stdv = sqrtf(fmaxf(var, 0.0f));
    float thr  = 3.0f * stdv;               // OUTLIER_THRESHOLD * std
    __syncthreads();                        // r1[0]/r2[0] consumed; safe to reuse

    float c[8];
    float amax = 0.f;
#pragma unroll
    for (int i = 0; i < 8; ++i) {
        c[i] = fminf(fmaxf(v[i], -thr), thr);
        amax = fmaxf(amax, fabsf(c[i]));
    }
    r1[t] = amax;
    __syncthreads();
    for (int off = 128; off > 0; off >>= 1) {
        if (t < off) r1[t] = fmaxf(r1[t], r1[t + off]);
        __syncthreads();
    }
    float scale = fmaxf(r1[0] / 127.0f, 1e-6f);   // QMAX = 127, EPS = 1e-6
    float inv_scale = 1.0f / scale;

    float o[8];
#pragma unroll
    for (int i = 0; i < 8; ++i) {
        float q = rintf(c[i] * inv_scale) * scale;     // RNE, matches jnp.round
        o[i] = (fabsf(v[i]) > thr) ? v[i] : q;
    }
    float4* op = (float4*)(row + t * 8);
    op[0] = make_float4(o[0], o[1], o[2], o[3]);
    op[1] = make_float4(o[4], o[5], o[6], o[7]);
}

extern "C" void kernel_launch(void* const* d_in, const int* in_sizes, int n_in,
                              void* d_out, int out_size, void* d_ws, size_t ws_size,
                              hipStream_t stream) {
    const float* x = (const float*)d_in[0];   // [B*S, D_in] fp32
    const float* W = (const float*)d_in[1];   // [D_out, D_in] fp32
    const float* b = (const float*)d_in[2];   // [D_out] fp32
    float* y = (float*)d_out;                 // [B*S, D_out] fp32

    const int D_out = in_sizes[2];            // 2048
    const int D_in  = in_sizes[1] / D_out;    // 2048
    const int M     = in_sizes[0] / D_in;     // 16384

    dim3 grid(D_out / TILE_N, M / TILE_M);    // (16, 128)
    oal_gemm_bf16x3_wmma<<<grid, 256, 0, stream>>>(x, W, b, y, M, D_out, D_in);
    oal_quant_rows<<<M, 256, 0, stream>>>(y, D_out);
}
